// AttnDecoder_38568806318758
// MI455X (gfx1250) — matrix operand order repaired and checked by reading, not compiled, not run on hardware
//
#include <hip/hip_runtime.h>
#include <hip/hip_bf16.h>
#include <math.h>

// ---------------- Problem constants ----------------
#define B_  128
#define T_  1024
#define I_  256
#define H_  512
#define V_  32000
#define KLSTM (I_ + H_ + H_)   // 1280 : [x | op | hp] vs [W_ih | W_hh]

// ---------------- WMMA fragment types (gfx1250, wave32) ----------------
typedef __attribute__((ext_vector_type(16))) __bf16 v16bf;
typedef __attribute__((ext_vector_type(8)))  __bf16 v8bf;
typedef __attribute__((ext_vector_type(8)))  float  v8f;

// TDM descriptor vector types (amdgpu-toolchain clang-23: 6-arg builtin form)
typedef __attribute__((ext_vector_type(4))) unsigned int uint32x4;
typedef __attribute__((ext_vector_type(8))) int          int32x8;
typedef __attribute__((ext_vector_type(4))) int          int32x4;

// A-fragment (16x32 bf16, MxK): lane<16 holds K = k0+{0..7,16..23},
// lane>=16 holds K = k0+{8..15,24..31}; row M = lane&15.
static __device__ __forceinline__ v16bf load_afrag_f32row(const float* __restrict__ row, int kb) {
  v16bf a;
#pragma unroll
  for (int j = 0; j < 8; ++j) {
    a[j]     = (__bf16)row[kb + j];
    a[8 + j] = (__bf16)row[kb + 16 + j];
  }
  return a;
}

// B-fragment (32x16 bf16, KxN): lane<16 holds K = k0+{0..7,16..23}, lane>=16 K = k0+{8..15,24..31};
// column N = lane&15.  Source = two runs of 8 f32 of weight row n (p, p+16).
static __device__ __forceinline__ v16bf load_bfrag_f32(const float* __restrict__ p) {
  v16bf b;
#pragma unroll
  for (int j = 0; j < 16; ++j) b[j] = (__bf16)p[j + ((j >> 3) << 3)];
  return b;
}

// Same, but source already bf16: two 16-byte vector loads.
static __device__ __forceinline__ v16bf load_bfrag_bf16(const __bf16* __restrict__ p) {
  v8bf lo = *(const v8bf*)(p);
  v8bf hi = *(const v8bf*)(p + 16);
  v16bf b;
#pragma unroll
  for (int j = 0; j < 8; ++j) { b[j] = lo[j]; b[8 + j] = hi[j]; }
  return b;
}

// ---------------- Generic GEMM:  C[M,N] = A[M,K] * W[N,K]^T  (bf16 WMMA, f32 acc)
// block = 128 threads (4 waves); block tile = 16(M) x 64(N); wave tile = 16x16.
template <int EPI>   // 0 = none, 1 = tanh
__global__ __launch_bounds__(128) void gemm_bf16_wmma(
    const float* __restrict__ A, const float* __restrict__ W, float* __restrict__ C,
    int M, int N, int K) {
  const int lane  = threadIdx.x & 31;
  const int wave  = threadIdx.x >> 5;
  const int tileM = blockIdx.y * 16;
  const int tileN = blockIdx.x * 64 + wave * 16;
  const int mrow  = tileM + (lane & 15);
  const int ncol  = tileN + (lane & 15);
  const int kbA   = (lane >> 4) * 8;
  const int kbB   = (lane >> 4) * 8;
  const float* arow = A + (size_t)mrow * K;
  const float* wrow = W + (size_t)ncol * K;

  v8f acc = {};
  for (int k0 = 0; k0 < K; k0 += 32) {
    v16bf a = load_afrag_f32row(arow, k0 + kbA);
    v16bf b = load_bfrag_f32(wrow + k0 + kbB);
    acc = __builtin_amdgcn_wmma_f32_16x16x32_bf16(false, a, false, b, (short)0, acc,
                                                  false, false);
  }
  // C layout: VGPR r -> row tileM + r (+8 for lanes 16..31), col = lane&15
  const int mbase = tileM + ((lane >> 4) * 8);
#pragma unroll
  for (int r = 0; r < 8; ++r) {
    float v = acc[r];
    if (EPI == 1) v = tanhf(v);
    C[(size_t)(mbase + r) * N + ncol] = v;
  }
}

// ---------------- Fused attention scores:
// scores[b,t] = Wa . tanh(q[b] + enc[b,t,:] @ Wm^T), without materializing k.
// One block = 16 (b,t) rows.  The 16x512 f32 encoder tile is DMA'd into LDS by the
// Tensor Data Mover (one TENSOR_LOAD_TO_LDS per block, TENSORcnt-tracked), with
// descriptor-driven LDS padding (4 dwords every 256) so fragment reads spread
// across banks.  4 waves x 8 N-tiles of bf16 WMMA, epilogue fused.
#define ROWPITCH 520   // 512 data dwords + 2 x 4 pad dwords per row
static __device__ __forceinline__ int pad_idx(int r, int c) {
  return r * ROWPITCH + c + ((c >> 8) << 2);   // +4 dwords after dword 256
}

__global__ __launch_bounds__(128) void attn_scores_kernel(
    const float* __restrict__ enc, const __bf16* __restrict__ Wm_bf,
    const float* __restrict__ q, const float* __restrict__ Wa,
    float* __restrict__ scores) {
  __shared__ __align__(16) float tileF[16 * ROWPITCH];   // 33.3 KB padded f32 tile
  __shared__ float red[4][32][8];

  const int bt0  = blockIdx.x * 16;      // flattened (b*T + t) row base, T|16
  const int b    = bt0 >> 10;            // T_ = 1024
  const int lane = threadIdx.x & 31;
  const int wave = threadIdx.x >> 5;

  const float* encBase = enc + (size_t)bt0 * H_;

#if __has_builtin(__builtin_amdgcn_tensor_load_to_lds)
  if (threadIdx.x == 0) {
    const unsigned long long ga = (unsigned long long)(uintptr_t)encBase;
    const unsigned int lds_addr = (unsigned int)(uintptr_t)&tileF[0];
    uint32x4 g0;
    g0[0] = 1u;                                    // count=1, user descriptor
    g0[1] = lds_addr;                              // lds_addr [63:32]
    g0[2] = (unsigned int)(ga & 0xFFFFFFFFu);      // global_addr [95:64]
    g0[3] = (unsigned int)((ga >> 32) & 0x1FFFFFFu) | (2u << 30);  // addr hi | type=2
    int32x8 g1;
    g1[0] = (2 << 16)        // data_size = 4B
          | (1 << 20)        // pad_enable
          | (7 << 22)        // pad_interval: 256 dwords
          | (3 << 25);       // pad_amount: 4 dwords
    g1[1] = (H_ & 0xFFFF) << 16;                   // tensor_dim0 = 512 (lo16)
    g1[2] = (16 << 16);                            // tensor_dim0 hi | tensor_dim1 = 16
    g1[3] = (H_ << 16);                            // tensor_dim1 hi | tile_dim0 = 512
    g1[4] = 16;                                    // tile_dim1 = 16, tile_dim2 = 0
    g1[5] = H_;                                    // tensor_dim0_stride = 512 (lo32)
    g1[6] = 0;                                     // stride hi | dim1_stride lo
    g1[7] = 0;
    int32x4 gz  = {0, 0, 0, 0};
    int32x8 gz8 = {0, 0, 0, 0, 0, 0, 0, 0};
    // clang-23 / therock-10.0 arity: (g0, g1, g2, g3, extra, cpol)
    __builtin_amdgcn_tensor_load_to_lds(g0, g1, gz, gz, gz8, 0);
  }
  __builtin_amdgcn_s_wait_tensorcnt(0);            // s_wait_tensorcnt 0
  __syncthreads();
#else
  for (int i = threadIdx.x; i < 16 * H_; i += 128) {
    int r = i >> 9, c = i & (H_ - 1);
    tileF[pad_idx(r, c)] = encBase[(size_t)r * H_ + c];
  }
  __syncthreads();
#endif

  const int arow = lane & 15;
  const int kbA  = (lane >> 4) * 8;
  const int kbB  = (lane >> 4) * 8;

  float partial[8];
#pragma unroll
  for (int r = 0; r < 8; ++r) partial[r] = 0.f;

  for (int nt = 0; nt < 8; ++nt) {
    const int n = (wave * 8 + nt) * 16 + (lane & 15);
    const __bf16* wrow = Wm_bf + (size_t)n * H_;
    v8f acc = {};
    for (int k0 = 0; k0 < H_; k0 += 32) {
      const int c0 = k0 + kbA;
      const float* p0 = &tileF[pad_idx(arow, c0)];
      const float* p1 = &tileF[pad_idx(arow, c0 + 16)];
      v16bf a;
#pragma unroll
      for (int j = 0; j < 8; ++j) { a[j] = (__bf16)p0[j]; a[8 + j] = (__bf16)p1[j]; }
      v16bf bf = load_bfrag_bf16(wrow + k0 + kbB);
      acc = __builtin_amdgcn_wmma_f32_16x16x32_bf16(false, a, false, bf, (short)0, acc,
                                                    false, false);
    }
    const float qv  = q[(size_t)b * H_ + n];
    const float wav = Wa[n];
#pragma unroll
    for (int r = 0; r < 8; ++r) partial[r] += tanhf(qv + acc[r]) * wav;
  }

#pragma unroll
  for (int r = 0; r < 8; ++r) red[wave][lane][r] = partial[r];
  __syncthreads();

  if (threadIdx.x < 16) {
    const int m  = threadIdx.x;
    const int r  = m & 7;
    const int l0 = (m < 8) ? 0 : 16;
    float s = 0.f;
    for (int w = 0; w < 4; ++w)
      for (int l = 0; l < 16; ++l) s += red[w][l0 + l][r];
    scores[bt0 + m] = s;
  }
}

// ---------------- Small elementwise / reduction kernels ----------------
__global__ __launch_bounds__(256) void wm_to_bf16_kernel(
    const float* __restrict__ Wm, __bf16* __restrict__ out) {
  int idx = blockIdx.x * 256 + threadIdx.x;           // H_*H_
  if (idx < H_ * H_) out[idx] = (__bf16)Wm[idx];
}

__global__ __launch_bounds__(256) void concat_lstm_in_kernel(
    const float* __restrict__ x, const float* __restrict__ op,
    const float* __restrict__ hp, float* __restrict__ out) {
  int idx = blockIdx.x * 256 + threadIdx.x;           // B_*KLSTM
  if (idx >= B_ * KLSTM) return;
  int b = idx / KLSTM, c = idx - b * KLSTM;
  float v;
  if (c < I_)            v = x[(size_t)b * I_ + c];
  else if (c < I_ + H_)  v = op[(size_t)b * H_ + (c - I_)];
  else                   v = hp[(size_t)b * H_ + (c - I_ - H_)];
  out[idx] = v;
}

__global__ __launch_bounds__(256) void concat_W_kernel(
    const float* __restrict__ W_ih, const float* __restrict__ W_hh,
    float* __restrict__ out) {
  int idx = blockIdx.x * 256 + threadIdx.x;           // 4H * KLSTM
  if (idx >= 4 * H_ * KLSTM) return;
  int r = idx / KLSTM, c = idx - r * KLSTM;
  out[idx] = (c < I_ + H_) ? W_ih[(size_t)r * (I_ + H_) + c]
                           : W_hh[(size_t)r * H_ + (c - I_ - H_)];
}

__global__ __launch_bounds__(256) void lstm_act_kernel(
    const float* __restrict__ gates, const float* __restrict__ b_ih,
    const float* __restrict__ b_hh, const float* __restrict__ cp,
    float* __restrict__ hn, float* __restrict__ cn) {
  int idx = blockIdx.x * 256 + threadIdx.x;           // B_*H_
  int b = idx >> 9, h = idx & (H_ - 1);
  const float* g = gates + (size_t)b * 4 * H_;
  float gi = g[h]            + b_ih[h]            + b_hh[h];
  float gf = g[H_ + h]       + b_ih[H_ + h]       + b_hh[H_ + h];
  float gg = g[2 * H_ + h]   + b_ih[2 * H_ + h]   + b_hh[2 * H_ + h];
  float go = g[3 * H_ + h]   + b_ih[3 * H_ + h]   + b_hh[3 * H_ + h];
  float ig = 1.f / (1.f + __expf(-gi));
  float fg = 1.f / (1.f + __expf(-gf));
  float gv = tanhf(gg);
  float og = 1.f / (1.f + __expf(-go));
  float c  = fg * cp[idx] + ig * gv;
  cn[idx] = c;
  hn[idx] = og * tanhf(c);
}

__global__ __launch_bounds__(256) void softmax_kernel(
    const float* __restrict__ scores, float* __restrict__ weights) {
  __shared__ float sm[256];
  const int b = blockIdx.x;
  const float* row = scores + (size_t)b * T_;
  float m = -1e30f;
  for (int t = threadIdx.x; t < T_; t += 256) m = fmaxf(m, row[t]);
  sm[threadIdx.x] = m; __syncthreads();
  for (int s = 128; s > 0; s >>= 1) {
    if (threadIdx.x < s) sm[threadIdx.x] = fmaxf(sm[threadIdx.x], sm[threadIdx.x + s]);
    __syncthreads();
  }
  m = sm[0]; __syncthreads();
  float sum = 0.f;
  for (int t = threadIdx.x; t < T_; t += 256) sum += __expf(row[t] - m);
  sm[threadIdx.x] = sum; __syncthreads();
  for (int s = 128; s > 0; s >>= 1) {
    if (threadIdx.x < s) sm[threadIdx.x] += sm[threadIdx.x + s];
    __syncthreads();
  }
  const float inv = 1.f / sm[0];
  for (int t = threadIdx.x; t < T_; t += 256)
    weights[(size_t)b * T_ + t] = __expf(row[t] - m) * inv;
}

__global__ __launch_bounds__(256) void context_kernel(
    const float* __restrict__ weights, const float* __restrict__ enc,
    float* __restrict__ ctx) {
  const int b = blockIdx.x;
  const int h = threadIdx.x;                      // 2 columns per thread
  const float* e = enc + (size_t)b * T_ * H_;
  const float* w = weights + (size_t)b * T_;
  float s0 = 0.f, s1 = 0.f;
  for (int t = 0; t < T_; ++t) {
    __builtin_prefetch(e + (size_t)(t + 8) * H_ + h, 0, 1);
    float wt = w[t];
    s0 += wt * e[(size_t)t * H_ + h];
    s1 += wt * e[(size_t)t * H_ + h + 256];
  }
  ctx[(size_t)b * H_ + h]       = s0;
  ctx[(size_t)b * H_ + h + 256] = s1;
}

__global__ __launch_bounds__(256) void concat_oin_kernel(
    const float* __restrict__ hn, const float* __restrict__ ctx,
    float* __restrict__ out) {
  int idx = blockIdx.x * 256 + threadIdx.x;       // B_ * 2H_
  if (idx >= B_ * 2 * H_) return;
  int b = idx / (2 * H_), c = idx - b * 2 * H_;
  out[idx] = (c < H_) ? hn[(size_t)b * H_ + c] : ctx[(size_t)b * H_ + (c - H_)];
}

__global__ __launch_bounds__(256) void log_softmax_kernel(float* __restrict__ out) {
  __shared__ float sm[256];
  const int b = blockIdx.x;
  float* row = out + (size_t)b * V_;
  float m = -1e30f;
  for (int v = threadIdx.x; v < V_; v += 256) m = fmaxf(m, row[v]);
  sm[threadIdx.x] = m; __syncthreads();
  for (int s = 128; s > 0; s >>= 1) {
    if (threadIdx.x < s) sm[threadIdx.x] = fmaxf(sm[threadIdx.x], sm[threadIdx.x + s]);
    __syncthreads();
  }
  m = sm[0]; __syncthreads();
  float sum = 0.f;
  for (int v = threadIdx.x; v < V_; v += 256) sum += __expf(row[v] - m);
  sm[threadIdx.x] = sum; __syncthreads();
  for (int s = 128; s > 0; s >>= 1) {
    if (threadIdx.x < s) sm[threadIdx.x] += sm[threadIdx.x + s];
    __syncthreads();
  }
  const float lse = m + __logf(sm[0]);
  for (int v = threadIdx.x; v < V_; v += 256) row[v] -= lse;
}

// ---------------- Host-side launcher ----------------
extern "C" void kernel_launch(void* const* d_in, const int* in_sizes, int n_in,
                              void* d_out, int out_size, void* d_ws, size_t ws_size,
                              hipStream_t stream) {
  const float* x    = (const float*)d_in[0];
  const float* hp   = (const float*)d_in[1];
  const float* cp   = (const float*)d_in[2];
  const float* op   = (const float*)d_in[3];
  const float* enc  = (const float*)d_in[4];
  const float* W_ih = (const float*)d_in[5];
  const float* W_hh = (const float*)d_in[6];
  const float* b_ih = (const float*)d_in[7];
  const float* b_hh = (const float*)d_in[8];
  const float* Wq   = (const float*)d_in[9];
  const float* Wm   = (const float*)d_in[10];
  const float* Wa   = (const float*)d_in[11];
  const float* Wo   = (const float*)d_in[12];
  const float* Wout = (const float*)d_in[13];

  float* out = (float*)d_out;
  // d_out tuple layout: output(B,V) | hn(1,B,H) | cn(1,B,H) | on(B,H) | weights(B,1,T)
  float* out_logits  = out;
  float* out_hn      = out + (size_t)B_ * V_;
  float* out_cn      = out_hn + (size_t)B_ * H_;
  float* out_on      = out_cn + (size_t)B_ * H_;
  float* out_weights = out_on + (size_t)B_ * H_;

  float* ws = (float*)d_ws;
  float* ws_lstm_in = ws;                                    // 128*1280
  float* ws_Wcat    = ws_lstm_in + (size_t)B_ * KLSTM;       // 2048*1280
  float* ws_gates   = ws_Wcat + (size_t)4 * H_ * KLSTM;      // 128*2048
  float* ws_q       = ws_gates + (size_t)B_ * 4 * H_;        // 128*512
  float* ws_scores  = ws_q + (size_t)B_ * H_;                // 128*1024
  float* ws_ctx     = ws_scores + (size_t)B_ * T_;           // 128*512
  float* ws_oin     = ws_ctx + (size_t)B_ * H_;              // 128*1024
  __bf16* ws_Wm_bf  = (__bf16*)(ws_oin + (size_t)B_ * 2 * H_); // 512*512 bf16

  // 1) concat [x | op | hp] and [W_ih | W_hh]; convert Wm to bf16 (L2-resident 512KB)
  concat_lstm_in_kernel<<<(B_ * KLSTM + 255) / 256, 256, 0, stream>>>(x, op, hp, ws_lstm_in);
  concat_W_kernel<<<(4 * H_ * KLSTM + 255) / 256, 256, 0, stream>>>(W_ih, W_hh, ws_Wcat);
  wm_to_bf16_kernel<<<(H_ * H_) / 256, 256, 0, stream>>>(Wm, ws_Wm_bf);

  // 2) LSTM gates GEMM: (128 x 2048), K = 1280
  gemm_bf16_wmma<0><<<dim3(4 * H_ / 64, B_ / 16), 128, 0, stream>>>(
      ws_lstm_in, ws_Wcat, ws_gates, B_, 4 * H_, KLSTM);

  // 3) LSTM activations -> hn, cn (written straight into d_out)
  lstm_act_kernel<<<(B_ * H_) / 256, 256, 0, stream>>>(ws_gates, b_ih, b_hh, cp,
                                                       out_hn, out_cn);

  // 4) q = hn @ Wq^T : (128 x 512), K = 512
  gemm_bf16_wmma<0><<<dim3(H_ / 64, B_ / 16), 128, 0, stream>>>(
      out_hn, Wq, ws_q, B_, H_, H_);

  // 5) fused k-projection + additive-attention scores (68.7 GFLOP, TDM-staged)
  attn_scores_kernel<<<(B_ * T_) / 16, 128, 0, stream>>>(enc, ws_Wm_bf, ws_q, Wa,
                                                         ws_scores);

  // 6) softmax over T -> weights (d_out)
  softmax_kernel<<<B_, 256, 0, stream>>>(ws_scores, out_weights);

  // 7) context = weights . enc  (second, final pass over encoder: 268 MB @ 23.3 TB/s)
  context_kernel<<<B_, 256, 0, stream>>>(out_weights, enc, ws_ctx);

  // 8) o_in = [hn | context]
  concat_oin_kernel<<<(B_ * 2 * H_ + 255) / 256, 256, 0, stream>>>(out_hn, ws_ctx, ws_oin);

  // 9) on = tanh(o_in @ Wo^T) : (128 x 512), K = 1024 (tanh epilogue, -> d_out)
  gemm_bf16_wmma<1><<<dim3(H_ / 64, B_ / 16), 128, 0, stream>>>(
      ws_oin, Wo, out_on, B_, H_, 2 * H_);

  // 10) logits = on @ Wout^T : (128 x 32000), K = 512 (-> d_out, Wout streamed f32)
  gemm_bf16_wmma<0><<<dim3(V_ / 64, B_ / 16), 128, 0, stream>>>(
      out_on, Wout, out_logits, B_, V_, H_);

  // 11) in-place log_softmax over V
  log_softmax_kernel<<<B_, 256, 0, stream>>>(out_logits);
}
